// BatchChildSumTreeLSTM_44925357916241
// MI455X (gfx1250) — compile-verified, matrix operand order checked by reading, not run on hardware
//
#include <hip/hip_runtime.h>
#include <hip/hip_bf16.h>

typedef __attribute__((ext_vector_type(16))) __bf16 v16bf;
typedef __attribute__((ext_vector_type(8)))  __bf16 v8bf;
typedef __attribute__((ext_vector_type(8)))  float  v8f;

#define HID     128
#define N_NODES 741376
#define N_ROOT  4096
#define WSTRIDE 136                    // padded LDS row stride (halfs): 68 dwords -> conflict-free
#define W_LDS   (8 * 128 * WSTRIDE)    // staged weights region (elements)
#define LDS_BYTES ((W_LDS + 8 * 16 * WSTRIDE) * 2)

// Load 16 bf16 for one lane of an A/B operand: two 16-byte chunks at p and p+16 elems.
__device__ __forceinline__ v16bf load_pair(const __bf16* p) {
  v8bf lo = *(const v8bf*)p;
  v8bf hi = *(const v8bf*)(p + 16);
  return __builtin_shufflevector(lo, hi, 0,1,2,3,4,5,6,7,8,9,10,11,12,13,14,15);
}

__device__ __forceinline__ v8f wmma_bf16(v16bf a, v16bf b, v8f c) {
  // 8 args: (neg_a, A, neg_b, B, c_mod, C, reuse_a, reuse_b)
  return __builtin_amdgcn_wmma_f32_16x16x32_bf16(false, a, false, b, (short)0, c, false, false);
}

__device__ __forceinline__ v8f sigmoid8(v8f x) {
  v8f r;
#pragma unroll
  for (int v = 0; v < 8; ++v) r[v] = 1.0f / (1.0f + __expf(-x[v]));
  return r;
}

// ---- Prep: transpose + convert weights to bf16, slot order [ix,ih,ox,oh,ux,uh,fx,fh] ----
__global__ void prep_weights(const float* Wix, const float* Wih, const float* Wox, const float* Woh,
                             const float* Wux, const float* Wuh, const float* Wfx, const float* Wfh,
                             __bf16* Wt) {
  int idx  = blockIdx.x * blockDim.x + threadIdx.x;   // 8*128*128 threads
  int slot = idx >> 14;
  int n    = (idx >> 7) & 127;
  int k    = idx & 127;
  const float* W;
  switch (slot) {
    case 0: W = Wix; break; case 1: W = Wih; break;
    case 2: W = Wox; break; case 3: W = Woh; break;
    case 4: W = Wux; break; case 5: W = Wuh; break;
    case 6: W = Wfx; break; default: W = Wfh; break;
  }
  Wt[idx] = (__bf16)W[k * HID + n];   // Wt[slot][n][k] = W[k][n]
}

// ---- Prep: embedding gather -> bf16 x features ----
__global__ void prep_x(const int* __restrict__ sen, const float* __restrict__ emb,
                       __bf16* __restrict__ xb) {
  int idx = blockIdx.x * blockDim.x + threadIdx.x;    // N_NODES*128 threads
  int row = idx >> 7;
  int k   = idx & 127;
  xb[idx] = (__bf16)emb[sen[row] * HID + k];
}

// ---- One tree level. 8 waves/block; each wave owns a 16-node tile.
//      All 8 weight matrices staged in padded LDS (bank-conflict-free B loads). ----
__global__ void __launch_bounds__(256)
tlstm_level(const __bf16* __restrict__ xb, __bf16* __restrict__ hb, float* __restrict__ cf,
            const __bf16* __restrict__ Wt,
            const float* __restrict__ b_ix, const float* __restrict__ b_ih,
            const float* __restrict__ b_ox, const float* __restrict__ b_oh,
            const float* __restrict__ b_ux, const float* __restrict__ b_uh,
            const float* __restrict__ b_fx, const float* __restrict__ b_fh,
            int o0, int c0, int R) {
  extern __shared__ __align__(16) __bf16 smem[];
  const int tid  = threadIdx.x;
  const int wave = tid >> 5;
  const int lane = tid & 31;
  const int g    = lane >> 4;      // lane group (selects K sub-chunks / M half)
  const int mlan = lane & 15;      // A row / C column within tile
  const int tile = blockIdx.x * 8 + wave;
  const int nodeBase = tile * 16;  // local node index of tile row m=0
  __bf16* hsum = smem + W_LDS + wave * (16 * WSTRIDE);

  // ---- Stage weights into LDS: 8*128 rows of 128 halfs -> padded rows of WSTRIDE ----
#pragma unroll 1
  for (int it = 0; it < 64; ++it) {
    int c    = it * 256 + tid;      // 16384 16-byte chunks total
    int rowg = c >> 4;              // slot*128 + n
    int ch   = c & 15;
    *(v8bf*)(smem + rowg * WSTRIDE + ch * 8) = *(const v8bf*)(Wt + rowg * 128 + ch * 8);
  }
  __syncthreads();

  // ---- h_sum over children -> LDS (bf16, padded 16 x WSTRIDE) ----
  if (R > 0) {
#pragma unroll 1
    for (int it = 0; it < 64; ++it) {
      int idx = it * 32 + lane;     // 2048 elements over 32 lanes
      int m = idx >> 7;
      int k = idx & 127;
      float s = 0.f;
      int crow = c0 + (nodeBase + m) * R;
      for (int r = 0; r < R; ++r)
        s += (float)hb[(size_t)(crow + r) * HID + k];
      hsum[m * WSTRIDE + k] = (__bf16)s;
    }
  }
  __syncthreads();

  const __bf16* xrow = xb + (size_t)(o0 + nodeBase + mlan) * HID;

#pragma unroll 1
  for (int nt = 0; nt < 8; ++nt) {
    const int n = nt * 16 + mlan;        // output column (C/D layout: n = lane&15)
    const __bf16* wb = smem + n * WSTRIDE + g * 8;   // within-slot B base for this lane
    const float bi  = b_ix[n] + b_ih[n];
    const float bo  = b_ox[n] + b_oh[n];
    const float bu  = b_ux[n] + b_uh[n];
    const float bfv = b_fx[n] + b_fh[n];
    v8f acc_i, acc_o, acc_u, pre_f;
#pragma unroll
    for (int v = 0; v < 8; ++v) { acc_i[v] = bi; acc_o[v] = bo; acc_u[v] = bu; pre_f[v] = bfv; }

#pragma unroll
    for (int ks = 0; ks < 4; ++ks) {
      v16bf ax = load_pair(xrow + ks * 32 + g * 8);
      acc_i = wmma_bf16(ax, load_pair(wb + 0 * (128 * WSTRIDE) + ks * 32), acc_i);  // W_ix
      acc_o = wmma_bf16(ax, load_pair(wb + 2 * (128 * WSTRIDE) + ks * 32), acc_o);  // W_ox
      acc_u = wmma_bf16(ax, load_pair(wb + 4 * (128 * WSTRIDE) + ks * 32), acc_u);  // W_ux
      pre_f = wmma_bf16(ax, load_pair(wb + 6 * (128 * WSTRIDE) + ks * 32), pre_f);  // W_fx
      if (R > 0) {
        v16bf ahs = load_pair(hsum + mlan * WSTRIDE + ks * 32 + g * 8);
        acc_i = wmma_bf16(ahs, load_pair(wb + 1 * (128 * WSTRIDE) + ks * 32), acc_i); // W_ih
        acc_o = wmma_bf16(ahs, load_pair(wb + 3 * (128 * WSTRIDE) + ks * 32), acc_o); // W_oh
        acc_u = wmma_bf16(ahs, load_pair(wb + 5 * (128 * WSTRIDE) + ks * 32), acc_u); // W_uh
      }
    }

    v8f fc = {};
    if (R > 0) {
#pragma unroll 1
      for (int r = 0; r < R; ++r) {
        v8f accf = pre_f;
        const __bf16* hrow = hb + (size_t)(c0 + (nodeBase + mlan) * R + r) * HID;
#pragma unroll
        for (int ks = 0; ks < 4; ++ks) {
          accf = wmma_bf16(load_pair(hrow + ks * 32 + g * 8),
                           load_pair(wb + 7 * (128 * WSTRIDE) + ks * 32),             // W_fh
                           accf);
        }
        accf = sigmoid8(accf);
#pragma unroll
        for (int v = 0; v < 8; ++v) {
          int m = g * 8 + v;                                      // C/D layout row
          float cc = cf[(size_t)(c0 + (nodeBase + m) * R + r) * HID + n];
          fc[v] += accf[v] * cc;
        }
      }
    }

    acc_i = sigmoid8(acc_i);
    acc_o = sigmoid8(acc_o);
#pragma unroll
    for (int v = 0; v < 8; ++v) {
      int m = g * 8 + v;
      float u    = tanhf(acc_u[v]);
      float cval = acc_i[v] * u + fc[v];
      float hval = acc_o[v] * tanhf(cval);
      size_t off = (size_t)(o0 + nodeBase + m) * HID + n;
      cf[off] = cval;
      hb[off] = (__bf16)hval;
    }
  }
}

// ---- Output projection: root_h (4096x128, bf16) @ W_out (128x4) + b_out ----
__global__ void out_proj(const __bf16* __restrict__ hb, const float* __restrict__ Wout,
                         const float* __restrict__ bout, float* __restrict__ out) {
  int idx  = blockIdx.x * blockDim.x + threadIdx.x;   // 4096*4 threads
  int node = idx >> 2;
  int cls  = idx & 3;
  float s = bout[cls];
#pragma unroll 4
  for (int k = 0; k < HID; ++k)
    s += (float)hb[node * HID + k] * Wout[k * 4 + cls];
  out[idx] = s;
}

extern "C" void kernel_launch(void* const* d_in, const int* in_sizes, int n_in,
                              void* d_out, int out_size, void* d_ws, size_t ws_size,
                              hipStream_t stream) {
  const int*   sen   = (const int*)d_in[0];
  const float* emb   = (const float*)d_in[1];
  const float* W_ix  = (const float*)d_in[2];
  const float* b_ix  = (const float*)d_in[3];
  const float* W_ih  = (const float*)d_in[4];
  const float* b_ih  = (const float*)d_in[5];
  const float* W_fx  = (const float*)d_in[6];
  const float* b_fx  = (const float*)d_in[7];
  const float* W_fh  = (const float*)d_in[8];
  const float* b_fh  = (const float*)d_in[9];
  const float* W_ox  = (const float*)d_in[10];
  const float* b_ox  = (const float*)d_in[11];
  const float* W_oh  = (const float*)d_in[12];
  const float* b_oh  = (const float*)d_in[13];
  const float* W_ux  = (const float*)d_in[14];
  const float* b_ux  = (const float*)d_in[15];
  const float* W_uh  = (const float*)d_in[16];
  const float* b_uh  = (const float*)d_in[17];
  const float* W_out = (const float*)d_in[18];
  const float* b_out = (const float*)d_in[19];

  // Workspace layout
  __bf16* Wt = (__bf16*)d_ws;                        // 8 * 128 * 128 bf16 (transposed)
  __bf16* xb = Wt + 8 * 128 * 128;                   // N_NODES * 128 bf16
  __bf16* hb = xb + (size_t)N_NODES * HID;           // N_NODES * 128 bf16
  float*  cf = (float*)(hb + (size_t)N_NODES * HID); // N_NODES * 128 f32

  (void)hipFuncSetAttribute((const void*)tlstm_level,
                            hipFuncAttributeMaxDynamicSharedMemorySize, LDS_BYTES);

  prep_weights<<<(8 * 128 * 128) / 256, 256, 0, stream>>>(
      W_ix, W_ih, W_ox, W_oh, W_ux, W_uh, W_fx, W_fh, Wt);
  prep_x<<<(N_NODES * HID) / 256, 256, 0, stream>>>(sen, emb, xb);

  // {o0, c0, R, nL} per level, leaf -> root
  const int lv[6][4] = {
    {479232,      0, 0, 262144},  // L5 leaves
    {217088, 479232, 1, 262144},  // L4
    { 86016, 217088, 2, 131072},  // L3
    { 20480,  86016, 2,  65536},  // L2
    {  4096,  20480, 4,  16384},  // L1
    {     0,   4096, 4,   4096},  // L0 root
  };
  for (int i = 0; i < 6; ++i) {
    dim3 grid(lv[i][3] / 128);  // 8 waves/block * 16 nodes/wave
    tlstm_level<<<grid, 256, LDS_BYTES, stream>>>(
        xb, hb, cf, Wt,
        b_ix, b_ih, b_ox, b_oh, b_ux, b_uh, b_fx, b_fh,
        lv[i][0], lv[i][1], lv[i][2]);
  }

  out_proj<<<(N_ROOT * 4) / 256, 256, 0, stream>>>(hb, W_out, b_out, (float*)d_out);
}
